// HetGAT_7842610283187
// MI455X (gfx1250) — matrix operand order repaired and checked by reading, not compiled, not run on hardware
//
#include <hip/hip_runtime.h>
#include <hip/hip_bf16.h>
#include <math.h>

#define NEG_SLOPE 0.2f
#define NHEAD 4
#define DHEAD 32
#define HID 128

typedef __attribute__((ext_vector_type(16))) __bf16 v16bf;
typedef __attribute__((ext_vector_type(8)))  float  v8f;

union Frag16 { v16bf v; unsigned int u[8]; };

// ---------------------------------------------------------------------------
// GEMM: C[M,128](f32) = A[M,K](bf16, row-major) * B[K,128], with B supplied
// transposed as Bt[128,K](bf16, row-major) so both fragments load as packed
// 32-bit pairs. One 16x16 output tile per wave, 8 waves/block -> 16x128 tile.
// Fragment layouts per CDNA5 ISA 7.12.2 (wave32):
//   A 16x32 bf16 : lanes 0-15 row M=lane, K {0..7,16..23}; lanes 16-31 K {8..15,24..31}
//   B 32x16 bf16 : lanes 0-15 col N=lane, K {0..7,16..23}; lanes 16-31 K {8..15,24..31}
//   C 16x16 f32  : VGPR v -> row v (+8 for lanes 16-31), col = lane%16
// ---------------------------------------------------------------------------
__global__ __launch_bounds__(256)
void gemm_bf16_wmma(const __bf16* __restrict__ A,
                    const __bf16* __restrict__ Bt,
                    float* __restrict__ Cm, int M, int K) {
  const int wave = threadIdx.x >> 5;
  const int lane = threadIdx.x & 31;
  const int half = lane >> 4;
  const int lr   = lane & 15;
  const int row0 = blockIdx.x << 4;
  const int col0 = wave << 4;
  int arow = row0 + lr; if (arow >= M) arow = M - 1;
  const __bf16* Arow = A  + (size_t)arow * K;
  const __bf16* Brow = Bt + (size_t)(col0 + lr) * K;
  v8f acc = {};
  for (int k0 = 0; k0 < K; k0 += 32) {
    Frag16 fa, fb;
#pragma unroll
    for (int j = 0; j < 4; ++j)
      fa.u[j]   = *(const unsigned int*)(Arow + k0 + 2*j + 8*half);
#pragma unroll
    for (int j = 0; j < 4; ++j)
      fa.u[4+j] = *(const unsigned int*)(Arow + k0 + 16 + 2*j + 8*half);
#pragma unroll
    for (int j = 0; j < 4; ++j)
      fb.u[j]   = *(const unsigned int*)(Brow + k0 + 2*j + 8*half);
#pragma unroll
    for (int j = 0; j < 4; ++j)
      fb.u[4+j] = *(const unsigned int*)(Brow + k0 + 16 + 2*j + 8*half);
    acc = __builtin_amdgcn_wmma_f32_16x16x32_bf16(false, fa.v, false, fb.v,
                                                  (short)0, acc, false, false);
  }
#pragma unroll
  for (int v = 0; v < 8; ++v) {
    int r = row0 + v + 8*half;
    if (r < M) Cm[(size_t)r * HID + col0 + lr] = acc[v];
  }
}

// ---------------------------------------------------------------------------
__global__ void f32_to_bf16_k(const float* __restrict__ in, __bf16* __restrict__ out,
                              size_t n) {
  size_t i = (size_t)blockIdx.x * blockDim.x + threadIdx.x;
  if (i < n) out[i] = (__bf16)in[i];
}

// W[K,128] f32 row-major -> Wt[128,K] bf16 row-major (i.e. column-major B)
__global__ void transpose_to_bf16_k(const float* __restrict__ W, __bf16* __restrict__ Wt,
                                    int K) {
  int i = blockIdx.x * blockDim.x + threadIdx.x;
  if (i >= HID * K) return;
  int n = i / K, k = i - n * K;
  Wt[i] = (__bf16)W[(size_t)k * HID + n];
}

__global__ void add_bias_k(float* __restrict__ h, const float* __restrict__ b, int N) {
  int i = blockIdx.x * blockDim.x + threadIdx.x;
  if (i < N * HID) h[i] += b[i & (HID - 1)];
}

__global__ void fill_k(float* __restrict__ p, float v, size_t n) {
  size_t i = (size_t)blockIdx.x * blockDim.x + threadIdx.x;
  if (i < n) p[i] = v;
}

__global__ void copy_k(const float* __restrict__ a, float* __restrict__ o, size_t n) {
  size_t i = (size_t)blockIdx.x * blockDim.x + threadIdx.x;
  if (i < n) o[i] = a[i];
}

// el[n,h] = dot(z[n, h*32 : h*32+32], al[h])
__global__ void el_k(const float* __restrict__ z, const float* __restrict__ al,
                     float* __restrict__ el, int N) {
  int i = blockIdx.x * blockDim.x + threadIdx.x;
  if (i >= N * NHEAD) return;
  int n = i >> 2, h = i & 3;
  const float* zp = z + (size_t)n * HID + h * DHEAD;
  const float* ap = al + h * DHEAD;
  float s = 0.f;
#pragma unroll
  for (int d = 0; d < DHEAD; ++d) s += zp[d] * ap[d];
  el[i] = s;
}

// Wr[k,h] = sum_d W[k, h*32+d] * ar[h,d]   (folds z_dst GEMM into a 128x4 matvec)
__global__ void wr_k(const float* __restrict__ W, const float* __restrict__ ar,
                     float* __restrict__ Wr) {
  int i = blockIdx.x * blockDim.x + threadIdx.x;
  if (i >= HID * NHEAD) return;
  int k = i >> 2, h = i & 3;
  const float* ap = ar + h * DHEAD;
  float s = 0.f;
#pragma unroll
  for (int d = 0; d < DHEAD; ++d) s += W[(size_t)k * HID + h * DHEAD + d] * ap[d];
  Wr[i] = s;
}

// er[n,h] = dot(h_dst[n,:], Wr[:,h])
__global__ void er_k(const float* __restrict__ hdst, const float* __restrict__ Wr,
                     float* __restrict__ er, int N) {
  int i = blockIdx.x * blockDim.x + threadIdx.x;
  if (i >= N * NHEAD) return;
  int n = i >> 2, h = i & 3;
  const float* hp = hdst + (size_t)n * HID;
  float s = 0.f;
  for (int k = 0; k < HID; ++k) s += hp[k] * Wr[k * NHEAD + h];
  er[i] = s;
}

__device__ __forceinline__ float lrelu(float x) { return x > 0.f ? x : NEG_SLOPE * x; }

__device__ __forceinline__ void atomicMaxF(float* addr, float val) {
  unsigned int* a = (unsigned int*)addr;
  unsigned int cur = *a;
  while (__uint_as_float(cur) < val) {
    unsigned int assumed = cur;
    cur = atomicCAS(a, assumed, __float_as_uint(val));
    if (cur == assumed) break;
  }
}

__global__ void edge_max_k(const int* __restrict__ src, const int* __restrict__ dst,
                           const float* __restrict__ el, const float* __restrict__ er,
                           float* __restrict__ m, int E) {
  int i = blockIdx.x * blockDim.x + threadIdx.x;
  if (i >= E * NHEAD) return;
  int e = i >> 2, h = i & 3;
  int s = src[e], d = dst[e];
  atomicMaxF(&m[d * NHEAD + h], lrelu(el[s * NHEAD + h] + er[d * NHEAD + h]));
}

__global__ void m_fix_k(float* __restrict__ m, int n) {
  int i = blockIdx.x * blockDim.x + threadIdx.x;
  if (i < n && m[i] == -__builtin_inff()) m[i] = 0.f;
}

__global__ void edge_sum_k(const int* __restrict__ src, const int* __restrict__ dst,
                           const float* __restrict__ el, const float* __restrict__ er,
                           const float* __restrict__ m, float* __restrict__ sums, int E) {
  int i = blockIdx.x * blockDim.x + threadIdx.x;
  if (i >= E * NHEAD) return;
  int e = i >> 2, h = i & 3;
  int s = src[e], d = dst[e];
  float v = __expf(lrelu(el[s * NHEAD + h] + er[d * NHEAD + h]) - m[d * NHEAD + h]);
  atomicAdd(&sums[d * NHEAD + h], v);
}

// warp per edge, lane = feature within head; 128 f32 atomic adds/edge (L2-resident)
__global__ void edge_agg_k(const int* __restrict__ src, const int* __restrict__ dst,
                           const float* __restrict__ el, const float* __restrict__ er,
                           const float* __restrict__ m, const float* __restrict__ sums,
                           const float* __restrict__ z, float* __restrict__ agg, int E) {
  int e = blockIdx.x * (blockDim.x >> 5) + (threadIdx.x >> 5);
  int lane = threadIdx.x & 31;
  if (e >= E) return;
  int s = src[e], d = dst[e];
  const float* zp = z + (size_t)s * HID;
  float* ap = agg + (size_t)d * HID;
#pragma unroll
  for (int h = 0; h < NHEAD; ++h) {
    float a = __expf(lrelu(el[s * NHEAD + h] + er[d * NHEAD + h]) - m[d * NHEAD + h])
              / sums[d * NHEAD + h];
    atomicAdd(&ap[h * DHEAD + lane], a * zp[h * DHEAD + lane]);
  }
}

// rst = agg + h_dst + bias; optional ReLU; accumulate into dst-ntype output
__global__ void epilogue_k(const float* __restrict__ agg, const float* __restrict__ hdst,
                           const float* __restrict__ bias, float* __restrict__ outacc,
                           int N, int act) {
  int i = blockIdx.x * blockDim.x + threadIdx.x;
  if (i >= N * HID) return;
  float r = agg[i] + hdst[i] + bias[i & (HID - 1)];
  if (act) r = fmaxf(r, 0.f);
  outacc[i] += r;
}

// ---------------------------------------------------------------------------
static inline unsigned blocks_for(size_t n, unsigned t) {
  return (unsigned)((n + t - 1) / t);
}

extern "C" void kernel_launch(void* const* d_in, const int* in_sizes, int n_in,
                              void* d_out, int out_size, void* d_ws, size_t ws_size,
                              hipStream_t stream) {
  const float* feat_a  = (const float*)d_in[0];
  const float* feat_b  = (const float*)d_in[1];
  const float* lin_W_a = (const float*)d_in[2];
  const float* lin_b_a = (const float*)d_in[3];
  const float* lin_W_b = (const float*)d_in[4];
  const float* lin_b_b = (const float*)d_in[5];
  const float* Wall    = (const float*)d_in[6];   // [3,3,128,128]
  const float* attn_l  = (const float*)d_in[7];   // [3,3,4,32]
  const float* attn_r  = (const float*)d_in[8];
  const float* biasAll = (const float*)d_in[9];   // [3,3,128]
  const int* srcs[3] = {(const int*)d_in[10], (const int*)d_in[12], (const int*)d_in[14]};
  const int* dsts[3] = {(const int*)d_in[11], (const int*)d_in[13], (const int*)d_in[15]};

  const int NA = in_sizes[0] / 768;
  const int NB = in_sizes[1] / 384;
  const int NE = in_sizes[10];
  const size_t maxN = (size_t)(NA > NB ? NA : NB);

  // ---- workspace bump allocator -------------------------------------------
  char* ws = (char*)d_ws;
  size_t off = 0;
  auto alloc = [&](size_t bytes) -> char* {
    char* p = ws + off;
    off = (off + bytes + 255) & ~(size_t)255;
    return p;
  };
  float*  ha     = (float*)alloc((size_t)NA * HID * 4);
  float*  hb     = (float*)alloc((size_t)NB * HID * 4);
  float*  oa     = (float*)alloc((size_t)NA * HID * 4);
  float*  ob     = (float*)alloc((size_t)NB * HID * 4);
  __bf16* ha16   = (__bf16*)alloc((size_t)NA * HID * 2);
  __bf16* hb16   = (__bf16*)alloc((size_t)NB * HID * 2);
  __bf16* WtAll  = (__bf16*)alloc((size_t)9 * HID * HID * 2);
  __bf16* linWtA = (__bf16*)alloc((size_t)HID * 768 * 2);
  __bf16* linWtB = (__bf16*)alloc((size_t)HID * 384 * 2);
  float*  Wr     = (float*)alloc((size_t)HID * NHEAD * 4);
  float*  elb    = (float*)alloc(maxN * NHEAD * 4);
  float*  erb    = (float*)alloc(maxN * NHEAD * 4);
  float*  mb     = (float*)alloc(maxN * NHEAD * 4);
  float*  sb     = (float*)alloc(maxN * NHEAD * 4);
  // overlapped region: feat->bf16 staging (initial only) vs z/agg (layers only)
  size_t zBytes    = maxN * HID * 4;
  size_t featBytes = (size_t)NA * 768 * 2 + (size_t)NB * 384 * 2;
  char* U = alloc(featBytes > 2 * zBytes ? featBytes : 2 * zBytes);
  float*  z    = (float*)U;
  float*  agg  = (float*)(U + zBytes);
  __bf16* fA16 = (__bf16*)U;
  __bf16* fB16 = (__bf16*)(U + (size_t)NA * 768 * 2);

  const unsigned T = 256;

  // ---- weight preparation (bf16, transposed) ------------------------------
  f32_to_bf16_k<<<blocks_for((size_t)NA * 768, T), T, 0, stream>>>(feat_a, fA16, (size_t)NA * 768);
  f32_to_bf16_k<<<blocks_for((size_t)NB * 384, T), T, 0, stream>>>(feat_b, fB16, (size_t)NB * 384);
  transpose_to_bf16_k<<<blocks_for((size_t)HID * 768, T), T, 0, stream>>>(lin_W_a, linWtA, 768);
  transpose_to_bf16_k<<<blocks_for((size_t)HID * 384, T), T, 0, stream>>>(lin_W_b, linWtB, 384);
  for (int i = 0; i < 9; ++i)
    transpose_to_bf16_k<<<blocks_for((size_t)HID * HID, T), T, 0, stream>>>(
        Wall + (size_t)i * HID * HID, WtAll + (size_t)i * HID * HID, HID);

  // ---- HeteroLinear -------------------------------------------------------
  gemm_bf16_wmma<<<blocks_for(NA, 16), T, 0, stream>>>(fA16, linWtA, ha, NA, 768);
  add_bias_k<<<blocks_for((size_t)NA * HID, T), T, 0, stream>>>(ha, lin_b_a, NA);
  gemm_bf16_wmma<<<blocks_for(NB, 16), T, 0, stream>>>(fB16, linWtB, hb, NB, 384);
  add_bias_k<<<blocks_for((size_t)NB * HID, T), T, 0, stream>>>(hb, lin_b_b, NB);

  // ---- 3 GAT layers -------------------------------------------------------
  float* cur_a = ha; float* cur_b = hb;
  float* nxt_a = oa; float* nxt_b = ob;
  for (int l = 0; l < 3; ++l) {
    int act = (l < 2) ? 1 : 0;
    f32_to_bf16_k<<<blocks_for((size_t)NA * HID, T), T, 0, stream>>>(cur_a, ha16, (size_t)NA * HID);
    f32_to_bf16_k<<<blocks_for((size_t)NB * HID, T), T, 0, stream>>>(cur_b, hb16, (size_t)NB * HID);
    fill_k<<<blocks_for((size_t)NA * HID, T), T, 0, stream>>>(nxt_a, 0.f, (size_t)NA * HID);
    fill_k<<<blocks_for((size_t)NB * HID, T), T, 0, stream>>>(nxt_b, 0.f, (size_t)NB * HID);

    for (int t = 0; t < 3; ++t) {
      int et = l * 3 + t;
      const float*  Wf = Wall   + (size_t)et * HID * HID;
      const __bf16* Wt = WtAll  + (size_t)et * HID * HID;
      const float*  al = attn_l + (size_t)et * HID;
      const float*  ar = attn_r + (size_t)et * HID;
      const float*  bi = biasAll + (size_t)et * HID;

      const __bf16* s16; const float* hdst; float* outacc; int Ns, Nd;
      if (t == 0)      { s16 = ha16; Ns = NA; hdst = cur_b; outacc = nxt_b; Nd = NB; } // a->b
      else if (t == 1) { s16 = hb16; Ns = NB; hdst = cur_a; outacc = nxt_a; Nd = NA; } // b->a
      else             { s16 = ha16; Ns = NA; hdst = cur_a; outacc = nxt_a; Nd = NA; } // a->a

      gemm_bf16_wmma<<<blocks_for(Ns, 16), T, 0, stream>>>(s16, Wt, z, Ns, HID);
      el_k<<<blocks_for((size_t)Ns * NHEAD, T), T, 0, stream>>>(z, al, elb, Ns);
      wr_k<<<blocks_for((size_t)HID * NHEAD, T), T, 0, stream>>>(Wf, ar, Wr);
      er_k<<<blocks_for((size_t)Nd * NHEAD, T), T, 0, stream>>>(hdst, Wr, erb, Nd);

      fill_k<<<blocks_for((size_t)Nd * NHEAD, T), T, 0, stream>>>(mb, -__builtin_inff(), (size_t)Nd * NHEAD);
      fill_k<<<blocks_for((size_t)Nd * NHEAD, T), T, 0, stream>>>(sb, 0.f, (size_t)Nd * NHEAD);
      fill_k<<<blocks_for((size_t)Nd * HID, T), T, 0, stream>>>(agg, 0.f, (size_t)Nd * HID);

      edge_max_k<<<blocks_for((size_t)NE * NHEAD, T), T, 0, stream>>>(srcs[t], dsts[t], elb, erb, mb, NE);
      m_fix_k<<<blocks_for((size_t)Nd * NHEAD, T), T, 0, stream>>>(mb, Nd * NHEAD);
      edge_sum_k<<<blocks_for((size_t)NE * NHEAD, T), T, 0, stream>>>(srcs[t], dsts[t], elb, erb, mb, sb, NE);
      edge_agg_k<<<blocks_for(NE, T / 32), T, 0, stream>>>(srcs[t], dsts[t], elb, erb, mb, sb, z, agg, NE);
      epilogue_k<<<blocks_for((size_t)Nd * HID, T), T, 0, stream>>>(agg, hdst, bi, outacc, Nd, act);
    }
    float* tmp;
    tmp = cur_a; cur_a = nxt_a; nxt_a = tmp;
    tmp = cur_b; cur_b = nxt_b; nxt_b = tmp;
  }

  // ---- output: concat [ha ; hb] -------------------------------------------
  float* out = (float*)d_out;
  copy_k<<<blocks_for((size_t)NA * HID, T), T, 0, stream>>>(cur_a, out, (size_t)NA * HID);
  copy_k<<<blocks_for((size_t)NB * HID, T), T, 0, stream>>>(cur_b, out + (size_t)NA * HID, (size_t)NB * HID);

  (void)n_in; (void)out_size; (void)ws_size;
}
